// SoftmaxSelfAttention_3040836845869
// MI455X (gfx1250) — compile-verified, matrix operand order checked by reading, not run on hardware
//
#include <hip/hip_runtime.h>
#include <hip/hip_bf16.h>

typedef __attribute__((ext_vector_type(16))) __bf16 v16bf;
typedef __attribute__((ext_vector_type(8)))  __bf16 v8bf;
typedef __attribute__((ext_vector_type(8)))  float  v8f;

// Problem sizes (from reference)
constexpr int Bb = 4, Hh = 16, Ss = 2048, Dd = 64;

// Tiling
constexpr int MT = 128;         // query rows per workgroup (8 waves x 16)
constexpr int KT = 32;          // keys per main-loop iteration
constexpr int NW = 8;           // waves per workgroup
constexpr int KSTR = Dd + 8;    // LDS stride (bf16) for K tile rows (key-major)
constexpr int VSTR = KT + 8;    // LDS stride (bf16) for V^T rows (d-major)
constexpr int PSTR = KT + 8;    // LDS stride (bf16) for per-wave P buffer
constexpr int SSTR = KT + 4;    // LDS stride (f32) for per-wave score buffer

constexpr float SCALE = 0.125f;               // 1/sqrt(64)
constexpr float LOG2E = 1.4426950408889634f;

__global__ __launch_bounds__(256, 1)
void fa_fwd_kernel(const float* __restrict__ Q, const float* __restrict__ K,
                   const float* __restrict__ V, const int* __restrict__ mask,
                   float* __restrict__ out)
{
    __shared__ alignas(16) __bf16 Kl[KT * KSTR];        // K tile, key-major (d contiguous)
    __shared__ alignas(16) __bf16 Vt[Dd * VSTR];        // V tile transposed (key contiguous)
    __shared__ alignas(16) __bf16 Pw[NW * 16 * PSTR];   // per-wave P (bf16, A-layout source)
    __shared__ alignas(16) float  Sw[NW * 16 * SSTR];   // per-wave raw scores (f32, row-major)
    __shared__ alignas(16) float  Fw[NW * 16];          // per-wave row broadcast line
    __shared__ float biasS[KT];                         // additive mask bias per key

    const int tid  = threadIdx.x;
    const int lane = tid & 31;
    const int wave = tid >> 5;
    const int half = lane >> 4;   // 16-lane half (selects K-run per ISA 7.12.2 layout)
    const int l15  = lane & 15;

    const int ntile = Ss / MT;                 // 16 query tiles per (b,h)
    const int bh    = blockIdx.x / ntile;      // b*H + h
    const int mtile = blockIdx.x % ntile;
    const int b     = bh / Hh;
    const int qbase = mtile * MT;

    const float* Qbh = Q + (size_t)bh * Ss * Dd;
    const float* Kbh = K + (size_t)bh * Ss * Dd;
    const float* Vbh = V + (size_t)bh * Ss * Dd;
    const int*   mb  = mask + (size_t)b * Ss;

    // ---- Q A-fragments (16x32 bf16): lane row = l15; runs K = 32f+8h+[0,8), 32f+16+8h+[0,8)
    v16bf qfrag[2];
    {
        const int    qrow = qbase + wave * 16 + l15;
        const float* qp   = Qbh + (size_t)qrow * Dd;
#pragma unroll
        for (int f = 0; f < 2; ++f) {
            const int d0 = 32 * f + 8 * half;
            const int d1 = 32 * f + 16 + 8 * half;
#pragma unroll
            for (int i = 0; i < 8; ++i) qfrag[f][i]     = (__bf16)qp[d0 + i];
#pragma unroll
            for (int i = 0; i < 8; ++i) qfrag[f][i + 8] = (__bf16)qp[d1 + i];
        }
    }

    // ---- software-pipelined K/V staging (regs <- global, LDS <- regs) ----
    const int skey = tid >> 3;         // 0..31: key within tile
    const int sd   = (tid & 7) * 8;    // 0..56: d-offset, 8 floats per thread
    float kst[8], vst[8];
    float bst = 0.0f;

    auto load_tile = [&](int t) {
        const int kb = t * KT;
        const float* kp = Kbh + (size_t)(kb + skey) * Dd + sd;
        const float* vp = Vbh + (size_t)(kb + skey) * Dd + sd;
#pragma unroll
        for (int i = 0; i < 8; ++i) kst[i] = kp[i];
#pragma unroll
        for (int i = 0; i < 8; ++i) vst[i] = vp[i];
        if (tid < KT) bst = (mb[kb + tid] == 0) ? -1.0e6f : 0.0f;
    };
    auto store_tile = [&]() {
#pragma unroll
        for (int i = 0; i < 8; ++i) Kl[skey * KSTR + sd + i] = (__bf16)kst[i];
#pragma unroll
        for (int i = 0; i < 8; ++i) Vt[(sd + i) * VSTR + skey] = (__bf16)vst[i];
        if (tid < KT) biasS[tid] = bst;
    };

    // ---- fragment builders: two contiguous 16B runs -> v16bf (2x ds_load_b128) ----
    auto mk_frag = [](const __bf16* p0, const __bf16* p1) -> v16bf {
        v8bf lo = *(const v8bf*)p0;
        v8bf hi = *(const v8bf*)p1;
        v16bf r;
#pragma unroll
        for (int i = 0; i < 8; ++i) { r[i] = lo[i]; r[i + 8] = hi[i]; }
        return r;
    };
    auto kfrag = [&](int kh, int f) -> v16bf {   // B-frag of K^T (contraction = d)
        const __bf16* base = &Kl[(kh * 16 + l15) * KSTR + 32 * f + 8 * half];
        return mk_frag(base, base + 16);
    };
    auto vfrag = [&](int c) -> v16bf {           // B-frag of V (contraction = key)
        const __bf16* base = &Vt[(c * 16 + l15) * VSTR + 8 * half];
        return mk_frag(base, base + 16);
    };
    auto pfrag = [&]() -> v16bf {                // A-frag of P (16 rows x 32 keys)
        const __bf16* base = &Pw[(wave * 16 + l15) * PSTR + 8 * half];
        return mk_frag(base, base + 16);
    };

    // ---- state ----
    v8f acc[4];                       // O accumulator, C layout: VGPR v -> row v+8*half
    v8f zero = {};
#pragma unroll
    for (int c = 0; c < 4; ++c) acc[c] = zero;
    float mrow = -3.0e38f;            // row-ownership layout: this lane owns row l15
    float lrow = 0.0f;                // (both halves hold identical copies)

    const int nT = Ss / KT;   // 64 key tiles
    load_tile(0);
    store_tile();
    __syncthreads();

    for (int t = 0; t < nT; ++t) {
        if (t + 1 < nT) load_tile(t + 1);   // next-tile global loads overlap compute

        // ---- S = Q K^T : two 16x16 score tiles, 64-d contraction in 2 WMMA steps ----
        v8f s0 = zero, s1 = zero;
#pragma unroll
        for (int f = 0; f < 2; ++f) {
            v16bf kb0 = kfrag(0, f);
            v16bf kb1 = kfrag(1, f);
            s0 = __builtin_amdgcn_wmma_f32_16x16x32_bf16(false, qfrag[f], false, kb0,
                                                         (short)0, s0, false, false);
            s1 = __builtin_amdgcn_wmma_f32_16x16x32_bf16(false, qfrag[f], false, kb1,
                                                         (short)0, s1, false, false);
        }

        // ---- scale + mask bias, spill scores (C layout -> row-major LDS) ----
        const float bias0 = biasS[l15];
        const float bias1 = biasS[16 + l15];
        {
            float* Smy = &Sw[wave * 16 * SSTR];
#pragma unroll
            for (int v = 0; v < 8; ++v) {
                const int row = v + 8 * half;
                Smy[row * SSTR + l15]      = s0[v] * SCALE + bias0;   // key = l15
                Smy[row * SSTR + 16 + l15] = s1[v] * SCALE + bias1;  // key = 16+l15
            }
        }
        asm volatile("s_wait_dscnt 0" ::: "memory");

        // ---- row-ownership online softmax: lane owns row l15, keys [16*half,16*half+16) ----
        float sv[16];
        {
            const float4* sp = (const float4*)&Sw[(wave * 16 + l15) * SSTR + 16 * half];
            float4 a = sp[0], c = sp[1], d = sp[2], e = sp[3];
            sv[0]=a.x; sv[1]=a.y; sv[2]=a.z; sv[3]=a.w;
            sv[4]=c.x; sv[5]=c.y; sv[6]=c.z; sv[7]=c.w;
            sv[8]=d.x; sv[9]=d.y; sv[10]=d.z; sv[11]=d.w;
            sv[12]=e.x; sv[13]=e.y; sv[14]=e.z; sv[15]=e.w;
        }
        float tmax[8];
#pragma unroll
        for (int i = 0; i < 8; ++i) tmax[i] = fmaxf(sv[i], sv[i + 8]);
#pragma unroll
        for (int s = 4; s >= 1; s >>= 1)
#pragma unroll
            for (int i = 0; i < 8; ++i)
                if (i < s) tmax[i] = fmaxf(tmax[i], tmax[i + s]);
        float rm = fmaxf(tmax[0], __shfl_xor(tmax[0], 16));   // combine the two halves

        const float mn  = fmaxf(mrow, rm);
        const float fac = __builtin_exp2f((mrow - mn) * LOG2E);
        mrow = mn;

        float p[16];
#pragma unroll
        for (int i = 0; i < 16; ++i) p[i] = __builtin_exp2f((sv[i] - mn) * LOG2E);
        float tsum[8];
#pragma unroll
        for (int i = 0; i < 8; ++i) tsum[i] = p[i] + p[i + 8];
#pragma unroll
        for (int s = 4; s >= 1; s >>= 1)
#pragma unroll
            for (int i = 0; i < 8; ++i)
                if (i < s) tsum[i] += tsum[i + s];
        const float rs = tsum[0] + __shfl_xor(tsum[0], 16);
        lrow = lrow * fac + rs;

        // broadcast fac to C layout + pack P to bf16 (2x ds_store_b128)
        if (half == 0) Fw[wave * 16 + l15] = fac;
        {
            v8bf pl, ph;
#pragma unroll
            for (int i = 0; i < 8; ++i) { pl[i] = (__bf16)p[i]; ph[i] = (__bf16)p[i + 8]; }
            __bf16* pd = &Pw[(wave * 16 + l15) * PSTR + 16 * half];
            *(v8bf*)pd       = pl;
            *(v8bf*)(pd + 8) = ph;
        }
        asm volatile("s_wait_dscnt 0" ::: "memory");

        // rescale accumulators with broadcast fac (rows v + 8*half)
        {
            const float4* fp = (const float4*)&Fw[wave * 16 + 8 * half];
            float4 fa = fp[0], fb = fp[1];
            float facv[8] = {fa.x, fa.y, fa.z, fa.w, fb.x, fb.y, fb.z, fb.w};
#pragma unroll
            for (int c = 0; c < 4; ++c)
#pragma unroll
                for (int v = 0; v < 8; ++v) acc[c][v] *= facv[v];
        }

        // ---- O += P V ----
        v16bf pa = pfrag();
#pragma unroll
        for (int c = 0; c < 4; ++c) {
            v16bf vb = vfrag(c);
            acc[c] = __builtin_amdgcn_wmma_f32_16x16x32_bf16(false, pa, false, vb,
                                                             (short)0, acc[c], false, false);
        }

        __syncthreads();                    // all waves done reading Kl/Vt
        if (t + 1 < nT) store_tile();       // regs -> LDS for next tile
        __syncthreads();
    }

    // ---- normalize (broadcast 1/l through LDS line) and write out ----
    if (half == 0) Fw[wave * 16 + l15] = 1.0f / lrow;
    asm volatile("s_wait_dscnt 0" ::: "memory");
    float invv[8];
    {
        const float4* fp = (const float4*)&Fw[wave * 16 + 8 * half];
        float4 fa = fp[0], fb = fp[1];
        invv[0]=fa.x; invv[1]=fa.y; invv[2]=fa.z; invv[3]=fa.w;
        invv[4]=fb.x; invv[5]=fb.y; invv[6]=fb.z; invv[7]=fb.w;
    }
    const int qrow0 = qbase + wave * 16;
    float* ob = out + (size_t)bh * Ss * Dd;
#pragma unroll
    for (int c = 0; c < 4; ++c)
#pragma unroll
        for (int v = 0; v < 8; ++v) {
            const int row = qrow0 + v + 8 * half;
            ob[(size_t)row * Dd + c * 16 + l15] = acc[c][v] * invv[v];
        }
}

extern "C" void kernel_launch(void* const* d_in, const int* in_sizes, int n_in,
                              void* d_out, int out_size, void* d_ws, size_t ws_size,
                              hipStream_t stream) {
    const float* Q    = (const float*)d_in[0];
    const float* K    = (const float*)d_in[1];
    const float* V    = (const float*)d_in[2];
    const int*   mask = (const int*)d_in[3];
    float*       out  = (float*)d_out;

    dim3 grid(Bb * Hh * (Ss / MT));   // 1024 workgroups
    dim3 block(256);                  // 8 wave32 waves
    fa_fwd_kernel<<<grid, block, 0, stream>>>(Q, K, V, mask, out);
}
